// ASE_Attention_Layer_79774722555994
// MI455X (gfx1250) — matrix operand order repaired and checked by reading, not compiled, not run on hardware
//
#include <hip/hip_runtime.h>
#include <hip/hip_bf16.h>

// ---------------- problem constants ----------------
#define N_NODES  50000
#define N_EDGES  800000
#define IN_DIM   256
#define N_HEADS  8
#define D_HEAD   32
#define QKV_COLS (3 * N_HEADS * D_HEAD)   // 768
#define NT_QKV   (QKV_COLS / 16)          // 48 column tiles
#define KSTEPS   (IN_DIM / 32)            // 8 K-steps of 32

typedef __attribute__((ext_vector_type(16))) _Float16 v16h;
typedef __attribute__((ext_vector_type(8)))  float    v8f;

// ---------------- workspace layout (bytes, 256B aligned) ----------------
// [0)          Wqkv swizzled f16 B-frags : 48*8*32*16*2 =    393216
// [393216)     Wedge swizzled f16 B-frags:  8*32*16*2   =      8192
// [401408)     qkv f32 [N][768]          :              = 153600000
// [154001408)  S12 f32 [E][16]           :              =  51200000
// [205201408)  Z   f32 [N][8]            :              =   1600000
#define WS_OFF_WQ   ((size_t)0)
#define WS_OFF_WE   ((size_t)393216)
#define WS_OFF_QKV  ((size_t)401408)
#define WS_OFF_S12  ((size_t)154001408)
#define WS_OFF_Z    ((size_t)205201408)

// ---------------- helpers ----------------
__device__ __forceinline__ float wave_sum32(float v) {
    #pragma unroll
    for (int m = 16; m >= 1; m >>= 1) v += __shfl_xor(v, m, 32);
    return v;
}

// load A fragment (16x32 f16 from f32 row-major, contiguous k = 16*half + s, same order as the B swizzle)
__device__ __forceinline__ v16h load_a_frag(const float* __restrict__ arow, int k0, int half) {
    const int abase = k0 + half * 16;
    float4 a0 = *(const float4*)(arow + abase);
    float4 a1 = *(const float4*)(arow + abase + 4);
    float4 a2 = *(const float4*)(arow + abase + 8);
    float4 a3 = *(const float4*)(arow + abase + 12);
    v16h af;
    af[0]=(_Float16)a0.x; af[1]=(_Float16)a0.y; af[2]=(_Float16)a0.z; af[3]=(_Float16)a0.w;
    af[4]=(_Float16)a1.x; af[5]=(_Float16)a1.y; af[6]=(_Float16)a1.z; af[7]=(_Float16)a1.w;
    af[8]=(_Float16)a2.x; af[9]=(_Float16)a2.y; af[10]=(_Float16)a2.z; af[11]=(_Float16)a2.w;
    af[12]=(_Float16)a3.x; af[13]=(_Float16)a3.y; af[14]=(_Float16)a3.z; af[15]=(_Float16)a3.w;
    return af;
}

// ---------------- kernel 1: swizzle Wqkv f32 -> f16 B-fragment layout ----------------
// dst[((nt*8 + ks)*32 + lane)*16 + j] = (f16) Wqkv[(ks*32 + (lane>>4)*16 + j)*768 + nt*16 + (lane&15)]
__global__ void k_swizzle_wqkv(const float* __restrict__ w, _Float16* __restrict__ dst) {
    int idx  = blockIdx.x * 256 + threadIdx.x;      // 768 blocks -> 196608
    int j    = idx & 15;
    int lane = (idx >> 4) & 31;
    int ks   = (idx >> 9) & 7;
    int nt   = idx >> 12;                           // 0..47
    int k = ks * 32 + ((lane >> 4) << 4) + j;
    int n = nt * 16 + (lane & 15);
    dst[idx] = (_Float16)w[k * QKV_COLS + n];
}

// ---------------- kernel 2: fold E1_w/E2_w and swizzle into B-fragment layout ----------
// column c<8 : sum_d E1_w[k, c*32+d]   (E.sum(-1) folded into the GEMM)
// column c>=8: E2_w[k, c-8]
__global__ void k_build_wedge_swz(const float* __restrict__ e1, const float* __restrict__ e2,
                                  _Float16* __restrict__ dst) {
    int idx  = blockIdx.x * 256 + threadIdx.x;      // 16 blocks -> 4096
    int j    = idx & 15;
    int lane = (idx >> 4) & 31;
    int ks   = idx >> 9;                            // 0..7
    int k = ks * 32 + ((lane >> 4) << 4) + j;
    int c = lane & 15;
    float acc;
    if (c < 8) {
        acc = 0.f;
        #pragma unroll
        for (int d = 0; d < D_HEAD; ++d) acc += e1[k * (N_HEADS * D_HEAD) + c * D_HEAD + d];
    } else {
        acc = e2[k * N_HEADS + (c - 8)];
    }
    dst[idx] = (_Float16)acc;
}

// ---------------- kernel 3: qkv = x @ Wqkv via WMMA (one 16x16 tile per wave) ----------
__global__ void k_qkv_gemm(const float* __restrict__ X, const _Float16* __restrict__ Wswz,
                           float* __restrict__ QKV) {
    const int wave = (blockIdx.x * blockDim.x + threadIdx.x) >> 5;   // 150000 waves
    const int lane = threadIdx.x & 31;
    const int mt = wave / NT_QKV;
    const int nt = wave % NT_QKV;
    const int r0 = mt * 16, c0 = nt * 16;
    const int half = lane >> 4, mr = lane & 15;

    const float* arow = X + (size_t)(r0 + mr) * IN_DIM;
    const v16h*  bp   = (const v16h*)Wswz + ((size_t)nt * KSTEPS) * 32 + lane;
    v8f acc = {};
    #pragma unroll
    for (int ks = 0; ks < KSTEPS; ++ks) {
        v16h a = load_a_frag(arow, ks * 32, half);
        v16h b = bp[ks * 32];                       // one aligned 32B fragment load
        acc = __builtin_amdgcn_wmma_f32_16x16x32_f16(false, a, false, b,
                                                     (short)0, acc, false, false);
    }
    float* drow = QKV + (size_t)(r0 + half * 8) * QKV_COLS + c0 + mr;
    #pragma unroll
    for (int v = 0; v < 8; ++v) drow[(size_t)v * QKV_COLS] = acc[v];
}

// ---------------- kernel 4: L2-normalize V rows (one wave per node*head) ----------------
__global__ void k_v_normalize(float* __restrict__ QKV, const float* __restrict__ vscale) {
    const int gwave = blockIdx.x * 8 + (threadIdx.x >> 5);   // 400000 waves
    const int lane  = threadIdx.x & 31;
    const int n = gwave >> 3, h = gwave & 7;
    float* vp = QKV + (size_t)n * QKV_COLS + h * (3 * D_HEAD) + 2 * D_HEAD;
    float v = vp[lane];
    float norm = sqrtf(wave_sum32(v * v));
    float s = vscale[0] / fmaxf(norm, 1e-12f);
    vp[lane] = v * s;
}

// ---------------- kernel 5: S12 = edge_attr @ Wedge (+bias on cols 8..15) via WMMA ------
__global__ void k_edge_gemm(const float* __restrict__ EA, const _Float16* __restrict__ Wswz,
                            const float* __restrict__ bias, float* __restrict__ S12) {
    const int wave = (blockIdx.x * blockDim.x + threadIdx.x) >> 5;   // 50000 waves
    const int lane = threadIdx.x & 31;
    const int r0 = wave * 16;
    const int half = lane >> 4, mr = lane & 15;

    const float* arow = EA + (size_t)(r0 + mr) * IN_DIM;
    const v16h*  bp   = (const v16h*)Wswz + lane;
    v8f acc = {};
    #pragma unroll
    for (int ks = 0; ks < KSTEPS; ++ks) {
        if (ks + 1 < KSTEPS)
            __builtin_prefetch(arow + (ks + 1) * 32 + (half ? 16 : 0), 0, 0);  // global_prefetch_b8
        v16h a = load_a_frag(arow, ks * 32, half);
        v16h b = bp[ks * 32];
        acc = __builtin_amdgcn_wmma_f32_16x16x32_f16(false, a, false, b,
                                                     (short)0, acc, false, false);
    }
    const float bv = (mr >= 8) ? bias[mr - 8] : 0.0f;
    float* drow = S12 + (size_t)(r0 + half * 8) * 16 + mr;
    #pragma unroll
    for (int v = 0; v < 8; ++v) drow[(size_t)v * 16] = acc[v] + bv;
}

// ---------------- kernel 6: per-edge attention + scatter-add (wave per edge) ------------
__global__ void k_edge_attn(const float* __restrict__ QKV, const float* __restrict__ S12,
                            const int* __restrict__ EI, float* __restrict__ wV,
                            float* __restrict__ Z) {
    const int e    = blockIdx.x * 8 + (threadIdx.x >> 5);    // 800000 waves
    const int lane = threadIdx.x & 31;
    const int s = EI[e];
    const int t = EI[N_EDGES + e];
    const float* qb = QKV + (size_t)t * QKV_COLS;   // Q rows of dst
    const float* kb = QKV + (size_t)s * QKV_COLS;   // K,V rows of src
    const float rs = 0.17677669529663687f;          // 1/sqrt(32)
    const float* srow = S12 + (size_t)e * 16;
    #pragma unroll
    for (int h = 0; h < N_HEADS; ++h) {
        float q = qb[h * 96 + lane];
        float k = kb[h * 96 + 32 + lane];
        float dot = wave_sum32(q * k);
        float sc = dot * rs * srow[h] + srow[8 + h];
        sc = fminf(fmaxf(sc, -8.0f), 8.0f);
        float ex = expf(sc);
        float v = kb[h * 96 + 64 + lane];
        atomicAdd(wV + (size_t)t * (N_HEADS * D_HEAD) + h * D_HEAD + lane, v * ex);
        if (lane == 0) atomicAdd(Z + (size_t)t * N_HEADS + h, ex);
    }
}

// ---------------- kernel 7: out = wV / (Z + 1e-6), in place ----------------
__global__ void k_finalize(float* __restrict__ out, const float* __restrict__ Z) {
    int idx = blockIdx.x * 256 + threadIdx.x;        // 50000 blocks
    int n = idx >> 8;
    int h = (idx & 255) >> 5;
    out[idx] = out[idx] / (Z[n * N_HEADS + h] + 1e-6f);
}

// ---------------- launcher ----------------
extern "C" void kernel_launch(void* const* d_in, const int* in_sizes, int n_in,
                              void* d_out, int out_size, void* d_ws, size_t ws_size,
                              hipStream_t stream) {
    const float* x      = (const float*)d_in[0];
    const int*   ei     = (const int*)  d_in[1];
    const float* ea     = (const float*)d_in[2];
    const float* wqkv   = (const float*)d_in[3];
    const float* vscale = (const float*)d_in[4];
    const float* e1     = (const float*)d_in[5];
    const float* e2     = (const float*)d_in[6];
    const float* e2b    = (const float*)d_in[7];
    float* out = (float*)d_out;

    char* ws = (char*)d_ws;
    _Float16* wqs = (_Float16*)(ws + WS_OFF_WQ);
    _Float16* wes = (_Float16*)(ws + WS_OFF_WE);
    float*    qkv = (float*)   (ws + WS_OFF_QKV);
    float*    s12 = (float*)   (ws + WS_OFF_S12);
    float*    z   = (float*)   (ws + WS_OFF_Z);

    // zero the scatter-add accumulators every call (graph-capture-safe memset nodes)
    hipMemsetAsync(out, 0, (size_t)N_NODES * N_HEADS * D_HEAD * sizeof(float), stream);
    hipMemsetAsync(z,   0, (size_t)N_NODES * N_HEADS * sizeof(float), stream);

    k_swizzle_wqkv   <<<QKV_COLS, 256, 0, stream>>>(wqkv, wqs);                // 196608 elems
    k_build_wedge_swz<<<16, 256, 0, stream>>>(e1, e2, wes);                    // 4096 elems
    k_qkv_gemm       <<<(N_NODES / 16) * NT_QKV / 8, 256, 0, stream>>>(x, wqs, qkv); // 18750
    k_v_normalize    <<<N_NODES, 256, 0, stream>>>(qkv, vscale);               // 400000 waves
    k_edge_gemm      <<<N_EDGES / 16 / 8, 256, 0, stream>>>(ea, wes, e2b, s12); // 6250
    k_edge_attn      <<<N_EDGES / 8, 256, 0, stream>>>(qkv, s12, ei, out, z);  // 100000
    k_finalize       <<<N_NODES, 256, 0, stream>>>(out, z);
}
